// Transformer_79809082294831
// MI455X (gfx1250) — matrix operand order repaired and checked by reading, not compiled, not run on hardware
//
#include <hip/hip_runtime.h>

typedef __bf16 bf16_t;
typedef __attribute__((ext_vector_type(16))) __bf16 v16bf;
typedef __attribute__((ext_vector_type(8)))  __bf16 v8bf;
typedef __attribute__((ext_vector_type(8)))  float  v8f;

#define D_   512
#define FF_  256
#define NE_  512
#define H_   8
#define E_   64
#define NROW 16384   // BS*NE

// ---------------------------------------------------------------------------
// CDNA5 wave32 WMMA bf16 16x16x32 fragment layouts (ISA 7.12.2):
// A (16x32, (m,k)): m = lane&15 ; two contiguous 16B runs at
//                   kk+((lane>>4)<<3) and +16.
// B (32x16, (k,n)): n = lane&15 ; two contiguous 16B runs of k at
//                   kk+((lane>>4)<<3) and +16 (requires k-contiguous storage,
//                   i.e. W pre-transposed to [n][k]).
// C/D f32 (16x16):  n = lane&15 ; m = i + ((lane>>4)<<3)
// ---------------------------------------------------------------------------
static __device__ __forceinline__ v16bf cat8(v8bf lo, v8bf hi) {
    return __builtin_shufflevector(lo, hi, 0, 1, 2, 3, 4, 5, 6, 7,
                                   8, 9, 10, 11, 12, 13, 14, 15);
}
static __device__ __forceinline__ v16bf ldfrag(const bf16_t* p, int step) {
    return cat8(*(const v8bf*)p, *(const v8bf*)(p + step));
}

// fp32 -> bf16 elementwise convert
__global__ void cvt_kernel(const float* __restrict__ in, bf16_t* __restrict__ out, int n) {
    int i = blockIdx.x * blockDim.x + threadIdx.x;
    if (i < n) out[i] = (bf16_t)in[i];
}

// fp32 (K x N) -> bf16 transposed (N x K)
__global__ void cvt_t_kernel(const float* __restrict__ in, bf16_t* __restrict__ out,
                             int K, int N) {
    int i = blockIdx.x * blockDim.x + threadIdx.x;
    if (i < K * N) {
        int k = i / N, n = i - k * N;
        out[(size_t)n * K + k] = (bf16_t)in[i];
    }
}

// ---------------------------------------------------------------------------
// Wave computes a 32x64 C strip (2 M-tiles x 4 N-tiles = 8 f32 accumulators).
// A fragments reused 4x, B fragments reused 2x; all operands via b128 loads,
// no LDS, no barriers. Block = 256 thr (8 waves along M => 256x64 tile).
// ---------------------------------------------------------------------------
template <int KDIM>
__device__ __forceinline__ void gemm_tile32x64(const bf16_t* __restrict__ A,
                                               const bf16_t* __restrict__ Wt, // [N][KDIM]
                                               int row0, int col0, int lane,
                                               v8f acc[2][4])
{
    for (int kk = 0; kk < KDIM; kk += 32) {
        const bf16_t* ap0 = A + (size_t)(row0 + (lane & 15)) * KDIM + kk + ((lane >> 4) << 3);
        const bf16_t* ap1 = ap0 + (size_t)16 * KDIM;
        if (kk + 32 < KDIM) __builtin_prefetch(ap0 + 32, 0, 0);   // global_prefetch_b8
        const v16bf a0 = ldfrag(ap0, 16);
        const v16bf a1 = ldfrag(ap1, 16);
#pragma unroll
        for (int nt = 0; nt < 4; ++nt) {
            const bf16_t* bp = Wt + (size_t)(col0 + nt * 16 + (lane & 15)) * KDIM
                                  + kk + ((lane >> 4) << 3);
            const v16bf bm = ldfrag(bp, 16);
            acc[0][nt] = __builtin_amdgcn_wmma_f32_16x16x32_bf16(false, a0, false, bm,
                                                                 (short)0, acc[0][nt], false, false);
            acc[1][nt] = __builtin_amdgcn_wmma_f32_16x16x32_bf16(false, a1, false, bm,
                                                                 (short)0, acc[1][nt], false, false);
        }
    }
}

// ---------------------------------------------------------------------------
// QKV: out = x @ W + b (M=16384,N=512,K=512). grid=(8,64,3), block=256.
// Q scaled by 1/8, layout [b][h][tok][64]; K layout [b][h][tok][64];
// V stored TRANSPOSED [b][h][emb][tok] so P@V B-fragments are contiguous.
// ---------------------------------------------------------------------------
__global__ void qkv_kernel(const bf16_t* __restrict__ xb,
                           const bf16_t* __restrict__ wqT, const bf16_t* __restrict__ wkT,
                           const bf16_t* __restrict__ wvT,
                           const float* __restrict__ bq, const float* __restrict__ bk,
                           const float* __restrict__ bv,
                           bf16_t* __restrict__ qo, bf16_t* __restrict__ ko,
                           bf16_t* __restrict__ vTo)
{
    const int lane = threadIdx.x & 31;
    const int wave = threadIdx.x >> 5;
    const int z    = blockIdx.z;
    const bf16_t* Wt   = (z == 0) ? wqT : (z == 1) ? wkT : wvT;
    const float*  bias = (z == 0) ? bq : (z == 1) ? bk : bv;
    const float   scale = (z == 0) ? 0.125f : 1.0f;

    const int row0 = blockIdx.y * 256 + wave * 32;
    const int col0 = blockIdx.x * 64;

    v8f acc[2][4] = {};
    gemm_tile32x64<D_>(xb, Wt, row0, col0, lane, acc);

#pragma unroll
    for (int mt = 0; mt < 2; ++mt) {
#pragma unroll
        for (int nt = 0; nt < 4; ++nt) {
#pragma unroll
            for (int i = 0; i < 8; ++i) {
                const int m   = mt * 16 + i + ((lane >> 4) << 3);
                const int col = col0 + nt * 16 + (lane & 15);
                const int tok = row0 + m;
                const float r = (acc[mt][nt][i] + bias[col]) * scale;
                const int bb = tok >> 9, tt = tok & (NE_ - 1);
                const int hh = col >> 6, e = col & (E_ - 1);
                const int bh = bb * H_ + hh;
                if (z == 0)      qo [((size_t)bh * NE_ + tt) * E_ + e] = (bf16_t)r;
                else if (z == 1) ko [((size_t)bh * NE_ + tt) * E_ + e] = (bf16_t)r;
                else             vTo[((size_t)bh * E_  + e) * NE_ + tt] = (bf16_t)r;
            }
        }
    }
}

// ---------------------------------------------------------------------------
// Attention: per (b*h, 32-query tile): S=QK^T (WMMA) -> fp32 softmax (8 thr/row)
// -> P@V (WMMA, P bf16 in LDS, V^T contiguous). grid=(16,256), block=256,
// dynamic LDS = 32*512*4 + 32*512*2 = 96 KB.
// ---------------------------------------------------------------------------
__global__ void attn_kernel(const bf16_t* __restrict__ qb,
                            const bf16_t* __restrict__ kb,
                            const bf16_t* __restrict__ vTb,
                            float* __restrict__ attended)
{
    extern __shared__ char smem[];
    float*  scores = (float*)smem;                    // [32][512]
    bf16_t* probs  = (bf16_t*)(smem + 32 * NE_ * 4);  // [32][512]
    __shared__ float red[32][9];

    const int lane = threadIdx.x & 31;
    const int wave = threadIdx.x >> 5;
    const int q0   = blockIdx.x * 32;
    const int bh   = blockIdx.y;
    const int bb   = bh >> 3, hh = bh & 7;
    const bf16_t* qh  = qb  + (size_t)bh * NE_ * E_;
    const bf16_t* kh  = kb  + (size_t)bh * NE_ * E_;
    const bf16_t* vTh = vTb + (size_t)bh * E_ * NE_;

    // ---- phase 1: S = Q K^T (2 x 32 tiles, 8 per wave), K = 64 ----
    for (int tt = wave; tt < 64; tt += 8) {
        const int mt = tt >> 5, nt = tt & 31;
        v8f acc = {};
#pragma unroll
        for (int kk = 0; kk < E_; kk += 32) {
            const bf16_t* ap = qh + (size_t)(q0 + mt * 16 + (lane & 15)) * E_ + kk + ((lane >> 4) << 3);
            const bf16_t* bp = kh + (size_t)(nt * 16 + (lane & 15)) * E_ + kk + ((lane >> 4) << 3);
            acc = __builtin_amdgcn_wmma_f32_16x16x32_bf16(false, ldfrag(ap, 16), false, ldfrag(bp, 16),
                                                          (short)0, acc, false, false);
        }
#pragma unroll
        for (int i = 0; i < 8; ++i) {
            const int m = i + ((lane >> 4) << 3);
            scores[(mt * 16 + m) * NE_ + nt * 16 + (lane & 15)] = acc[i];
        }
    }
    __syncthreads();

    // ---- phase 2: fp32 softmax, 8 threads per row ----
    {
        const int r = threadIdx.x >> 3, s = threadIdx.x & 7;
        float* row = scores + r * NE_ + s * 64;
        float mx = row[0];
        for (int j = 1; j < 64; ++j) mx = fmaxf(mx, row[j]);
        red[r][s] = mx;
        __syncthreads();
        if (s == 0) {
            float m2 = red[r][0];
#pragma unroll
            for (int i = 1; i < 8; ++i) m2 = fmaxf(m2, red[r][i]);
            red[r][8] = m2;
        }
        __syncthreads();
        const float rm = red[r][8];
        float sum = 0.0f;
        for (int j = 0; j < 64; ++j) { float e = __expf(row[j] - rm); row[j] = e; sum += e; }
        __syncthreads();
        red[r][s] = sum;
        __syncthreads();
        if (s == 0) {
            float t2 = red[r][0];
#pragma unroll
            for (int i = 1; i < 8; ++i) t2 += red[r][i];
            red[r][8] = 1.0f / t2;
        }
        __syncthreads();
        const float inv = red[r][8];
        bf16_t* prow = probs + r * NE_ + s * 64;
        for (int j = 0; j < 64; ++j) prow[j] = (bf16_t)(row[j] * inv);
    }
    __syncthreads();

    // ---- phase 3: O = P @ V (2 x 4 tiles, 1 per wave), K = 512 ----
    {
        const int mt = wave >> 2, nt = wave & 3;
        v8f acc = {};
        for (int kk = 0; kk < NE_; kk += 32) {
            const bf16_t* ap = probs + (size_t)(mt * 16 + (lane & 15)) * NE_ + kk + ((lane >> 4) << 3);
            const bf16_t* bp = vTh + (size_t)(nt * 16 + (lane & 15)) * NE_ + kk + ((lane >> 4) << 3);
            acc = __builtin_amdgcn_wmma_f32_16x16x32_bf16(false, ldfrag(ap, 16), false, ldfrag(bp, 16),
                                                          (short)0, acc, false, false);
        }
#pragma unroll
        for (int i = 0; i < 8; ++i) {
            const int m = i + ((lane >> 4) << 3);
            attended[(size_t)(bb * NE_ + q0 + mt * 16 + m) * D_ + hh * E_ + nt * 16 + (lane & 15)] = acc[i];
        }
    }
}

// ---------------------------------------------------------------------------
// Residual + LayerNorm over D=512. grid = 16384 rows, block = 256.
// ---------------------------------------------------------------------------
__global__ void ln_kernel(const float* __restrict__ a, const float* __restrict__ resid,
                          const float* __restrict__ g, const float* __restrict__ be,
                          float* __restrict__ outf, bf16_t* __restrict__ outb)
{
    __shared__ float red[256];
    const int row = blockIdx.x;
    const int t   = threadIdx.x;
    const size_t base = (size_t)row * D_;

    const float v0 = a[base + t]       + resid[base + t];
    const float v1 = a[base + 256 + t] + resid[base + 256 + t];

    red[t] = v0 + v1;
    __syncthreads();
    for (int s = 128; s > 0; s >>= 1) { if (t < s) red[t] += red[t + s]; __syncthreads(); }
    const float mu = red[0] * (1.0f / D_);
    __syncthreads();

    const float d0 = v0 - mu, d1 = v1 - mu;
    red[t] = d0 * d0 + d1 * d1;
    __syncthreads();
    for (int s = 128; s > 0; s >>= 1) { if (t < s) red[t] += red[t + s]; __syncthreads(); }
    const float rs = rsqrtf(red[0] * (1.0f / D_) + 1e-5f);

    const float o0 = d0 * rs * g[t] + be[t];
    const float o1 = d1 * rs * g[256 + t] + be[256 + t];
    outf[base + t] = o0;
    outf[base + 256 + t] = o1;
    if (outb) {
        outb[base + t] = (bf16_t)o0;
        outb[base + 256 + t] = (bf16_t)o1;
    }
}

// ---------------------------------------------------------------------------
// FF1: relu(h @ W1 + b1)  M=16384 N=256 K=512 -> bf16. grid=(4,64).
// ---------------------------------------------------------------------------
__global__ void ff1_kernel(const bf16_t* __restrict__ hb, const bf16_t* __restrict__ w1T,
                           const float* __restrict__ b1, bf16_t* __restrict__ f1)
{
    const int lane = threadIdx.x & 31, wave = threadIdx.x >> 5;
    const int row0 = blockIdx.y * 256 + wave * 32;
    const int col0 = blockIdx.x * 64;
    v8f acc[2][4] = {};
    gemm_tile32x64<D_>(hb, w1T, row0, col0, lane, acc);
#pragma unroll
    for (int mt = 0; mt < 2; ++mt)
#pragma unroll
        for (int nt = 0; nt < 4; ++nt)
#pragma unroll
            for (int i = 0; i < 8; ++i) {
                const int m = mt * 16 + i + ((lane >> 4) << 3);
                const int col = col0 + nt * 16 + (lane & 15);
                f1[(size_t)(row0 + m) * FF_ + col] = (bf16_t)fmaxf(acc[mt][nt][i] + b1[col], 0.0f);
            }
}

// ---------------------------------------------------------------------------
// FF2: f1 @ W2 + b2  M=16384 N=512 K=256 -> f32. grid=(8,64).
// ---------------------------------------------------------------------------
__global__ void ff2_kernel(const bf16_t* __restrict__ f1, const bf16_t* __restrict__ w2T,
                           const float* __restrict__ b2, float* __restrict__ out)
{
    const int lane = threadIdx.x & 31, wave = threadIdx.x >> 5;
    const int row0 = blockIdx.y * 256 + wave * 32;
    const int col0 = blockIdx.x * 64;
    v8f acc[2][4] = {};
    gemm_tile32x64<FF_>(f1, w2T, row0, col0, lane, acc);
#pragma unroll
    for (int mt = 0; mt < 2; ++mt)
#pragma unroll
        for (int nt = 0; nt < 4; ++nt)
#pragma unroll
            for (int i = 0; i < 8; ++i) {
                const int m = mt * 16 + i + ((lane >> 4) << 3);
                const int col = col0 + nt * 16 + (lane & 15);
                out[(size_t)(row0 + m) * D_ + col] = acc[mt][nt][i] + b2[col];
            }
}

// ---------------------------------------------------------------------------
extern "C" void kernel_launch(void* const* d_in, const int* in_sizes, int n_in,
                              void* d_out, int out_size, void* d_ws, size_t ws_size,
                              hipStream_t stream)
{
    const float* x   = (const float*)d_in[0];
    const float* Wq  = (const float*)d_in[1];
    const float* bq  = (const float*)d_in[2];
    const float* Wk  = (const float*)d_in[3];
    const float* bk  = (const float*)d_in[4];
    const float* Wv  = (const float*)d_in[5];
    const float* bv  = (const float*)d_in[6];
    const float* W1  = (const float*)d_in[7];
    const float* bf1 = (const float*)d_in[8];
    const float* W2  = (const float*)d_in[9];
    const float* bf2 = (const float*)d_in[10];
    const float* g1  = (const float*)d_in[11];
    const float* be1 = (const float*)d_in[12];
    const float* g2  = (const float*)d_in[13];
    const float* be2 = (const float*)d_in[14];

    char* ws = (char*)d_ws;
    size_t off = 0;
    auto take = [&](size_t bytes) { char* p = ws + off; off += (bytes + 255) & ~size_t(255); return p; };

    bf16_t* x_bf  = (bf16_t*)take((size_t)NROW * D_ * 2);
    bf16_t* wqT   = (bf16_t*)take((size_t)D_ * D_ * 2);
    bf16_t* wkT   = (bf16_t*)take((size_t)D_ * D_ * 2);
    bf16_t* wvT   = (bf16_t*)take((size_t)D_ * D_ * 2);
    bf16_t* w1T   = (bf16_t*)take((size_t)D_ * FF_ * 2);
    bf16_t* w2T   = (bf16_t*)take((size_t)FF_ * D_ * 2);
    bf16_t* q_bf  = (bf16_t*)take((size_t)NROW * D_ * 2);
    bf16_t* k_bf  = (bf16_t*)take((size_t)NROW * D_ * 2);
    bf16_t* vT_bf = (bf16_t*)take((size_t)NROW * D_ * 2);
    float*  attnd = (float*) take((size_t)NROW * D_ * 4);
    float*  h_f   = (float*) take((size_t)NROW * D_ * 4);
    bf16_t* h_bf  = (bf16_t*)take((size_t)NROW * D_ * 2);
    bf16_t* f1    = (bf16_t*)take((size_t)NROW * FF_ * 2);
    float*  f2    = (float*) take((size_t)NROW * D_ * 4);

    const dim3 blk(256);
    auto nb = [](int n) { return dim3((n + 255) / 256); };

    cvt_kernel  <<<nb(NROW * D_), blk, 0, stream>>>(x,  x_bf, NROW * D_);
    cvt_t_kernel<<<nb(D_ * D_),   blk, 0, stream>>>(Wq, wqT, D_, D_);
    cvt_t_kernel<<<nb(D_ * D_),   blk, 0, stream>>>(Wk, wkT, D_, D_);
    cvt_t_kernel<<<nb(D_ * D_),   blk, 0, stream>>>(Wv, wvT, D_, D_);
    cvt_t_kernel<<<nb(D_ * FF_),  blk, 0, stream>>>(W1, w1T, D_, FF_);
    cvt_t_kernel<<<nb(FF_ * D_),  blk, 0, stream>>>(W2, w2T, FF_, D_);

    qkv_kernel<<<dim3(D_ / 64, NROW / 256, 3), blk, 0, stream>>>(
        x_bf, wqT, wkT, wvT, bq, bk, bv, q_bf, k_bf, vT_bf);

    const size_t attn_lds = (size_t)32 * NE_ * 4 + (size_t)32 * NE_ * 2; // 96 KB
    attn_kernel<<<dim3(NE_ / 32, 256), blk, attn_lds, stream>>>(q_bf, k_bf, vT_bf, attnd);

    ln_kernel<<<dim3(NROW), blk, 0, stream>>>(attnd, x, g1, be1, h_f, h_bf);

    ff1_kernel<<<dim3(FF_ / 64, NROW / 256), blk, 0, stream>>>(h_bf, w1T, bf1, f1);
    ff2_kernel<<<dim3(D_ / 64, NROW / 256), blk, 0, stream>>>(f1, w2T, bf2, f2);

    ln_kernel<<<dim3(NROW), blk, 0, stream>>>(f2, h_f, g2, be2, (float*)d_out, nullptr);
}
